// MemoryEfficientAttention_26603027432179
// MI455X (gfx1250) — matrix operand order repaired and checked by reading, not compiled, not run on hardware
//
#include <hip/hip_runtime.h>
#include <hip/hip_bf16.h>

typedef _Float16 half_t;
typedef __attribute__((ext_vector_type(16))) _Float16 v16h;
typedef __attribute__((ext_vector_type(8)))  _Float16 v8h;
typedef __attribute__((ext_vector_type(4)))  _Float16 v4h;
typedef __attribute__((ext_vector_type(8)))  float    v8f;
typedef __attribute__((ext_vector_type(4)))  float    v4f;

union V16H { v16h v; v8h h[2]; };

#define S_LEN 4096
#define EDIM  512
#define HDIM  64
#define NH    8
#define MG    5000   // mask leading dimension (MAX_GENES)

// ---------------------------------------------------------------------------
// Tiled WMMA GEMM:  out[M,N] = (A[M,K] @ W[N,K]^T + bias[N]) * oscale
//   A_HALF:    A is f16 (else f32, converted while staging)
//   OUT_HALF:  output f16 (else f32)
//   OUT_TRANS: output stored transposed as out[N][M] (f16)
// Block: 256 threads (8 waves), tile 128x128, BK=32, waves arranged 4(M) x 2(N),
// each wave computes 32x64 = 2x4 WMMA 16x16 fragments.
// ---------------------------------------------------------------------------
template<bool A_HALF, bool OUT_HALF, bool OUT_TRANS>
__global__ __launch_bounds__(256) void gemm_wmma_f16(
    const void* __restrict__ Aptr, const float* __restrict__ W,
    const float* __restrict__ bias, void* __restrict__ outp,
    int M, int N, int K, float oscale)
{
    __shared__ alignas(16) half_t As[128 * 40];   // 128 rows x 32 cols, pad to 40
    __shared__ alignas(16) half_t Bs[128 * 40];

    const int t    = threadIdx.x;
    const int lane = t & 31;
    const int wv   = t >> 5;
    const int lg   = lane >> 4;       // lane group (0/1)
    const int ln   = lane & 15;
    const int wm   = wv >> 1;         // 0..3
    const int wn   = wv & 1;          // 0..1
    const int mbase = blockIdx.y * 128;
    const int nbase = blockIdx.x * 128;

    v8f c[2][4];
    #pragma unroll
    for (int mf = 0; mf < 2; ++mf)
        #pragma unroll
        for (int nf = 0; nf < 4; ++nf)
            #pragma unroll
            for (int r = 0; r < 8; ++r) c[mf][nf][r] = 0.0f;

    for (int kb = 0; kb < K; kb += 32) {
        // ---- stage A tile (128x32) into LDS as f16 ----
        if constexpr (A_HALF) {
            const half_t* A = (const half_t*)Aptr;
            #pragma unroll
            for (int j = 0; j < 2; ++j) {
                int lin = t + j * 256;            // 512 chunks of 8 halves
                int row = lin >> 2, c8 = lin & 3;
                *(v8h*)&As[row * 40 + c8 * 8] =
                    *(const v8h*)&A[(size_t)(mbase + row) * K + kb + c8 * 8];
            }
        } else {
            const float* A = (const float*)Aptr;
            #pragma unroll
            for (int j = 0; j < 4; ++j) {
                int lin = t + j * 256;            // 1024 chunks of 4 floats
                int row = lin >> 3, c4 = lin & 7;
                v4f f = *(const v4f*)&A[(size_t)(mbase + row) * K + kb + c4 * 4];
                v4h hh = { (half_t)f[0], (half_t)f[1], (half_t)f[2], (half_t)f[3] };
                *(v4h*)&As[row * 40 + c4 * 4] = hh;
            }
        }
        // ---- stage W tile (128x32), f32 -> f16 ----
        #pragma unroll
        for (int j = 0; j < 4; ++j) {
            int lin = t + j * 256;
            int row = lin >> 3, c4 = lin & 7;
            v4f f = *(const v4f*)&W[(size_t)(nbase + row) * K + kb + c4 * 4];
            v4h hh = { (half_t)f[0], (half_t)f[1], (half_t)f[2], (half_t)f[3] };
            *(v4h*)&Bs[row * 40 + c4 * 4] = hh;
        }
        __syncthreads();

        // ---- fragment loads + WMMA ----
        V16H a[2];
        #pragma unroll
        for (int mf = 0; mf < 2; ++mf) {
            int rl = wm * 32 + mf * 16 + ln;      // A row in tile
            a[mf].h[0] = *(const v8h*)&As[rl * 40 + lg * 8];
            a[mf].h[1] = *(const v8h*)&As[rl * 40 + 16 + lg * 8];
        }
        #pragma unroll
        for (int nf = 0; nf < 4; ++nf) {
            int cl = wn * 64 + nf * 16 + ln;      // W row (output col) in tile
            V16H b;
            b.h[0] = *(const v8h*)&Bs[cl * 40 + lg * 8];
            b.h[1] = *(const v8h*)&Bs[cl * 40 + 16 + lg * 8];
            #pragma unroll
            for (int mf = 0; mf < 2; ++mf)
                c[mf][nf] = __builtin_amdgcn_wmma_f32_16x16x32_f16(
                    false, a[mf].v, false, b.v, (short)0, c[mf][nf], false, false);
        }
        __syncthreads();
    }

    // ---- epilogue ----
    #pragma unroll
    for (int mf = 0; mf < 2; ++mf) {
        #pragma unroll
        for (int nf = 0; nf < 4; ++nf) {
            int n  = nbase + wn * 64 + nf * 16 + ln;
            float bv = bias[n];
            int m0 = mbase + wm * 32 + mf * 16 + 8 * lg;   // rows m0..m0+7
            if constexpr (OUT_TRANS) {
                half_t* Ot = (half_t*)outp;                 // [N][M]
                v8h pack;
                #pragma unroll
                for (int r = 0; r < 8; ++r) pack[r] = (half_t)((c[mf][nf][r] + bv) * oscale);
                *(v8h*)&Ot[(size_t)n * M + m0] = pack;
            } else if constexpr (OUT_HALF) {
                half_t* O = (half_t*)outp;                  // [M][N]
                #pragma unroll
                for (int r = 0; r < 8; ++r)
                    O[(size_t)(m0 + r) * N + n] = (half_t)((c[mf][nf][r] + bv) * oscale);
            } else {
                float* O = (float*)outp;                    // [M][N]
                #pragma unroll
                for (int r = 0; r < 8; ++r)
                    O[(size_t)(m0 + r) * N + n] = (c[mf][nf][r] + bv) * oscale;
            }
        }
    }
}

// ---------------------------------------------------------------------------
// Flash attention with sparse mask. Q is pre-scaled by 1/sqrt(D) = 0.125.
// Grid: (S/64)*NH blocks, 128 threads (4 waves). Wave w owns 16 query rows.
// Qh, Kh: f16 [S][E] (head slice contiguous). Vt: f16 [E][S] (transposed).
// Att: f16 [S][E].
// Online softmax anchored at m >= 0 so masked entries (-1e30) underflow to
// exactly 0 in __expf without a guard; fully-masked rows give l=0 -> output 0
// (matches where + softmax + nan_to_num in the reference).
// ---------------------------------------------------------------------------
__global__ __launch_bounds__(128) void attn_wmma(
    const half_t* __restrict__ Qh, const half_t* __restrict__ Kh,
    const half_t* __restrict__ Vt, const unsigned char* __restrict__ mask,
    half_t* __restrict__ Att)
{
    __shared__ alignas(16) half_t P_lds[4][16][72];

    const int w    = threadIdx.x >> 5;
    const int lane = threadIdx.x & 31;
    const int lg   = lane >> 4;
    const int ln   = lane & 15;
    const int h     = blockIdx.x % NH;
    const int qtile = blockIdx.x / NH;
    const int qbase = qtile * 64 + w * 16;

    // Q A-fragments (kept for whole kernel): 16 rows x 64 (two K=32 chunks)
    V16H qa[2];
    {
        const half_t* qrow = Qh + (size_t)(qbase + ln) * EDIM + h * HDIM;
        #pragma unroll
        for (int kc = 0; kc < 2; ++kc) {
            qa[kc].h[0] = *(const v8h*)(qrow + kc * 32 + lg * 8);
            qa[kc].h[1] = *(const v8h*)(qrow + kc * 32 + 16 + lg * 8);
        }
    }

    float m_run[8], l_run[8];
    v8f o[4];
    #pragma unroll
    for (int r = 0; r < 8; ++r) { m_run[r] = 0.0f; l_run[r] = 0.0f; }
    #pragma unroll
    for (int nf = 0; nf < 4; ++nf)
        #pragma unroll
        for (int r = 0; r < 8; ++r) o[nf][r] = 0.0f;

    for (int kb = 0; kb < S_LEN; kb += 64) {
        // ---- S = Q @ K^T for this 16x64 strip (Q pre-scaled) ----
        v8f s[4];
        #pragma unroll
        for (int nf = 0; nf < 4; ++nf)
            #pragma unroll
            for (int r = 0; r < 8; ++r) s[nf][r] = 0.0f;

        #pragma unroll
        for (int nf = 0; nf < 4; ++nf) {
            const half_t* krow = Kh + (size_t)(kb + nf * 16 + ln) * EDIM + h * HDIM;
            #pragma unroll
            for (int kc = 0; kc < 2; ++kc) {
                V16H b;
                b.h[0] = *(const v8h*)(krow + kc * 32 + lg * 8);
                b.h[1] = *(const v8h*)(krow + kc * 32 + 16 + lg * 8);
                s[nf] = __builtin_amdgcn_wmma_f32_16x16x32_f16(
                    false, qa[kc].v, false, b.v, (short)0, s[nf], false, false);
            }
        }

        // ---- mask (C-frag: row = qbase + r + 8*lg, col = kb + nf*16 + ln) ----
        float p[4][8];
        float rowmax[8];
        #pragma unroll
        for (int r = 0; r < 8; ++r) rowmax[r] = -1e30f;
        #pragma unroll
        for (int nf = 0; nf < 4; ++nf) {
            int col = kb + nf * 16 + ln;
            #pragma unroll
            for (int r = 0; r < 8; ++r) {
                int row = qbase + r + 8 * lg;
                float v = mask[(size_t)row * MG + col] ? s[nf][r] : -1e30f;
                p[nf][r] = v;
                rowmax[r] = fmaxf(rowmax[r], v);
            }
        }
        // reduce row max across the 16 lanes of each half-wave
        #pragma unroll
        for (int r = 0; r < 8; ++r)
            #pragma unroll
            for (int off = 1; off < 16; off <<= 1)
                rowmax[r] = fmaxf(rowmax[r], __shfl_xor(rowmax[r], off, 32));

        float factor[8], rowsum[8];
        #pragma unroll
        for (int r = 0; r < 8; ++r) {
            float mn = fmaxf(m_run[r], rowmax[r]);   // >= 0 always
            factor[r] = __expf(m_run[r] - mn);
            m_run[r] = mn;
            float rs = 0.0f;
            #pragma unroll
            for (int nf = 0; nf < 4; ++nf) {
                float pv = __expf(p[nf][r] - mn);    // masked: exp(-1e30-mn) == 0
                p[nf][r] = pv;
                rs += pv;
            }
            rowsum[r] = rs;
        }
        #pragma unroll
        for (int r = 0; r < 8; ++r) {
            #pragma unroll
            for (int off = 1; off < 16; off <<= 1)
                rowsum[r] += __shfl_xor(rowsum[r], off, 32);
            l_run[r] = l_run[r] * factor[r] + rowsum[r];
        }
        // rescale running output
        #pragma unroll
        for (int nf = 0; nf < 4; ++nf)
            #pragma unroll
            for (int r = 0; r < 8; ++r) o[nf][r] *= factor[r];

        // ---- reshape P: C-layout -> LDS -> A-layout ----
        #pragma unroll
        for (int nf = 0; nf < 4; ++nf)
            #pragma unroll
            for (int r = 0; r < 8; ++r)
                P_lds[w][r + 8 * lg][nf * 16 + ln] = (half_t)p[nf][r];
        // same-wave DS RAW: compiler inserts s_wait_dscnt

        // ---- O += P @ V ----
        #pragma unroll
        for (int kc = 0; kc < 2; ++kc) {
            V16H a;
            a.h[0] = *(const v8h*)&P_lds[w][ln][kc * 32 + lg * 8];
            a.h[1] = *(const v8h*)&P_lds[w][ln][kc * 32 + 16 + lg * 8];
            #pragma unroll
            for (int nf = 0; nf < 4; ++nf) {
                const half_t* vrow = Vt + (size_t)(h * HDIM + nf * 16 + ln) * S_LEN
                                        + kb + kc * 32;
                V16H b;
                b.h[0] = *(const v8h*)(vrow + lg * 8);
                b.h[1] = *(const v8h*)(vrow + 16 + lg * 8);
                o[nf] = __builtin_amdgcn_wmma_f32_16x16x32_f16(
                    false, a.v, false, b.v, (short)0, o[nf], false, false);
            }
        }
    }

    // ---- normalize + store (zero for fully-masked rows, matching nan_to_num) ----
    #pragma unroll
    for (int nf = 0; nf < 4; ++nf)
        #pragma unroll
        for (int r = 0; r < 8; ++r) {
            float inv = l_run[r] > 0.0f ? 1.0f / l_run[r] : 0.0f;
            Att[(size_t)(qbase + r + 8 * lg) * EDIM + h * HDIM + nf * 16 + ln] =
                (half_t)(o[nf][r] * inv);
        }
}

// ---------------------------------------------------------------------------
extern "C" void kernel_launch(void* const* d_in, const int* in_sizes, int n_in,
                              void* d_out, int out_size, void* d_ws, size_t ws_size,
                              hipStream_t stream)
{
    const float* x    = (const float*)d_in[0];
    const float* Wq   = (const float*)d_in[1];
    const float* bq   = (const float*)d_in[2];
    const float* Wk   = (const float*)d_in[3];
    const float* bk   = (const float*)d_in[4];
    const float* Wv   = (const float*)d_in[5];
    const float* bv   = (const float*)d_in[6];
    const float* Wo   = (const float*)d_in[7];
    const float* bo   = (const float*)d_in[8];
    const unsigned char* mask = (const unsigned char*)d_in[9];

    const size_t elems = (size_t)S_LEN * EDIM;          // 2M elements
    char* ws = (char*)d_ws;
    half_t* Qh  = (half_t*)(ws);                        //  4 MB  [S][E]  (pre-scaled by 0.125)
    half_t* Kh  = (half_t*)(ws + 2 * elems);            //  4 MB  [S][E]
    half_t* Vt  = (half_t*)(ws + 4 * elems);            //  4 MB  [E][S] (transposed)
    half_t* Att = (half_t*)(ws + 6 * elems);            //  4 MB  [S][E]

    dim3 ggrid(EDIM / 128, S_LEN / 128);                // (4, 32)
    // Q = (x @ Wq^T + bq) * 0.125  -> f16 [S][E]   (fold in 1/sqrt(HDIM), exact in f16)
    gemm_wmma_f16<false, true,  false><<<ggrid, 256, 0, stream>>>(x, Wq, bq, Qh, S_LEN, EDIM, EDIM, 0.125f);
    // K = x @ Wk^T + bk  -> f16 [S][E]
    gemm_wmma_f16<false, true,  false><<<ggrid, 256, 0, stream>>>(x, Wk, bk, Kh, S_LEN, EDIM, EDIM, 1.0f);
    // V = x @ Wv^T + bv  -> f16 transposed [E][S]
    gemm_wmma_f16<false, true,  true ><<<ggrid, 256, 0, stream>>>(x, Wv, bv, Vt, S_LEN, EDIM, EDIM, 1.0f);
    // masked flash attention
    attn_wmma<<<(S_LEN / 64) * NH, 128, 0, stream>>>(Qh, Kh, Vt, mask, Att);
    // out = Att @ Wo^T + bo -> f32 [S][E]
    gemm_wmma_f16<true,  false, false><<<ggrid, 256, 0, stream>>>(Att, Wo, bo, (float*)d_out, S_LEN, EDIM, EDIM, 1.0f);

    (void)in_sizes; (void)n_in; (void)out_size; (void)ws_size;
}
